// GCN_21131239096355
// MI455X (gfx1250) — matrix operand order repaired and checked by reading, not compiled, not run on hardware
//
#include <hip/hip_runtime.h>
#include <hip/hip_bf16.h>

typedef __attribute__((ext_vector_type(16))) __bf16    v16bf;
typedef __attribute__((ext_vector_type(8)))  float     v8f;
typedef __attribute__((ext_vector_type(8)))  unsigned  v8u;

union ABFrag { v8u u; v16bf bf; };

#define DD   256
#define LDP  264   // padded LDS row stride (ushorts): 132 dwords -> bank step 4/row

__device__ __forceinline__ unsigned short f2bf(float f) {
    unsigned u = __float_as_uint(f);
    u += 0x7FFFu + ((u >> 16) & 1u);   // round-to-nearest-even
    return (unsigned short)(u >> 16);
}

// ---------------- small utility kernels ----------------
__global__ void zero_u32_kernel(unsigned* p, int n) {
    int i = blockIdx.x * blockDim.x + threadIdx.x;
    if (i < n) p[i] = 0u;
}

__global__ void degree_kernel(const int* __restrict__ dst, unsigned* __restrict__ cnt, int E) {
    int i = blockIdx.x * blockDim.x + threadIdx.x;
    if (i < E) atomicAdd(&cnt[dst[i]], 1u);
}

__global__ void dinv_kernel(const unsigned* __restrict__ cnt, float* __restrict__ dinv, int N) {
    int i = blockIdx.x * blockDim.x + threadIdx.x;
    if (i < N) dinv[i] = rsqrtf((float)(cnt[i] + 1u));   // +1 for self-loop
}

// Wt[n*D + k] = bf16(W[k*D + n])  (transposed so B fragments are contiguous dwords)
__global__ void packW_kernel(const float* __restrict__ W, unsigned short* __restrict__ Wt) {
    int i = blockIdx.x * blockDim.x + threadIdx.x;
    if (i < DD * DD) {
        int n = i / DD, k = i % DD;
        Wt[i] = f2bf(W[k * DD + n]);
    }
}

// ---------------- fused LayerNorm + WMMA GEMM ----------------
// One block = 16 rows of x. 8 waves; wave w computes column tiles 2w, 2w+1.
__global__ void __launch_bounds__(256)
ln_gemm_kernel(const float* __restrict__ x,
               const float* __restrict__ gamma, const float* __restrict__ beta,
               const unsigned short* __restrict__ Wt,
               float* __restrict__ xw, int N)
{
    __shared__ unsigned short xn[16 * LDP];

    const int tid  = threadIdx.x;
    const int lane = tid & 31;
    const int wave = tid >> 5;
    const int rowbase = blockIdx.x * 16;

    // ---- LayerNorm: each wave normalizes rows 2w and 2w+1 into LDS (bf16) ----
    #pragma unroll
    for (int rr = 0; rr < 2; ++rr) {
        int row = 2 * wave + rr;
        int grow = min(rowbase + row, N - 1);
        const float* xr = x + (size_t)grow * DD;
        float v[8];
        float s = 0.f;
        #pragma unroll
        for (int i = 0; i < 8; ++i) { v[i] = xr[lane + 32 * i]; s += v[i]; }
        #pragma unroll
        for (int off = 16; off > 0; off >>= 1) s += __shfl_xor(s, off, 32);
        float mu = s * (1.0f / DD);
        float q = 0.f;
        #pragma unroll
        for (int i = 0; i < 8; ++i) { float d = v[i] - mu; q += d * d; }
        #pragma unroll
        for (int off = 16; off > 0; off >>= 1) q += __shfl_xor(q, off, 32);
        float rstd = rsqrtf(q * (1.0f / DD) + 1e-5f);
        #pragma unroll
        for (int i = 0; i < 8; ++i) {
            int col = lane + 32 * i;
            float y = (v[i] - mu) * rstd * gamma[col] + beta[col];
            xn[row * LDP + col] = f2bf(y);
        }
    }
    __syncthreads();

    // ---- GEMM: D = A(16x32 bf16) x B(32x16 bf16) + C, 8 K-steps ----
    v8f acc0 = {}, acc1 = {};
    const int m    = lane & 15;                  // A row / B col within tile
    const int aoff = (lane < 16) ? 0 : 16;       // A K-offset (contiguous, matches B)
    const int boff = (lane < 16) ? 0 : 16;       // B K-offset (contiguous, matches A)
    const int col0 = (2 * wave) * 16 + m;        // global output column, tile 0
    const int col1 = col0 + 16;                  // tile 1

    #pragma unroll
    for (int kk = 0; kk < DD; kk += 32) {
        ABFrag a, b0, b1;
        #pragma unroll
        for (int j = 0; j < 4; ++j) {
            a.u[j]     = *(const unsigned*)&xn[m * LDP + kk + aoff + 2 * j];
            a.u[j + 4] = *(const unsigned*)&xn[m * LDP + kk + aoff + 8 + 2 * j];
        }
        #pragma unroll
        for (int j = 0; j < 8; ++j) {
            b0.u[j] = *(const unsigned*)&Wt[(size_t)col0 * DD + kk + boff + 2 * j];
            b1.u[j] = *(const unsigned*)&Wt[(size_t)col1 * DD + kk + boff + 2 * j];
        }
        acc0 = __builtin_amdgcn_wmma_f32_16x16x32_bf16(false, a.bf, false, b0.bf,
                                                       (short)0, acc0, false, false);
        acc1 = __builtin_amdgcn_wmma_f32_16x16x32_bf16(false, a.bf, false, b1.bf,
                                                       (short)0, acc1, false, false);
    }

    // ---- store D tiles: VGPR r -> M = r + (lane<16?0:8), N = lane&15 ----
    const int mb = (lane < 16) ? 0 : 8;
    #pragma unroll
    for (int r = 0; r < 8; ++r) {
        int row = rowbase + r + mb;
        if (row < N) {
            xw[(size_t)row * DD + col0] = acc0[r];
            xw[(size_t)row * DD + col1] = acc1[r];
        }
    }
}

// ---------------- self-loop init: acc[n][d] = xw[n][d] * dinv[n]^2 ----------------
__global__ void selfloop_kernel(const float* __restrict__ xw, const float* __restrict__ dinv,
                                float* __restrict__ acc, int N)
{
    int i = blockIdx.x * blockDim.x + threadIdx.x;     // over N*D/4 float4 chunks
    int total = N * (DD / 4);
    if (i >= total) return;
    int n = i / (DD / 4);
    float dv = dinv[n]; dv *= dv;
    float4 v = ((const float4*)xw)[i];
    v.x *= dv; v.y *= dv; v.z *= dv; v.w *= dv;
    ((float4*)acc)[i] = v;
}

// ---------------- edge scatter: acc[dst] += xw[src] * dinv[src]*dinv[dst] ----------------
__global__ void scatter_kernel(const int* __restrict__ ei, const float* __restrict__ xw,
                               const float* __restrict__ dinv, float* __restrict__ acc, int E)
{
    int gid = blockIdx.x * blockDim.x + threadIdx.x;   // E * 64 work items (float4 each)
    if (gid >= E * (DD / 4)) return;
    int e = gid >> 6;
    int c = gid & 63;
    int src = ei[e];
    int dst = ei[E + e];
    float norm = dinv[src] * dinv[dst];
    float4 v = ((const float4*)(xw + (size_t)src * DD))[c];
    float* out = acc + (size_t)dst * DD + c * 4;
    unsafeAtomicAdd(out + 0, v.x * norm);
    unsafeAtomicAdd(out + 1, v.y * norm);
    unsafeAtomicAdd(out + 2, v.z * norm);
    unsafeAtomicAdd(out + 3, v.w * norm);
}

// ---------------- epilogue: out = relu(acc + b + x), in place on d_out ----------------
__global__ void epilogue_kernel(float* __restrict__ acc, const float* __restrict__ x,
                                const float* __restrict__ b, int N)
{
    int i = blockIdx.x * blockDim.x + threadIdx.x;
    int total = N * (DD / 4);
    if (i >= total) return;
    int c4 = i & (DD / 4 - 1);
    float4 a = ((float4*)acc)[i];
    float4 xv = ((const float4*)x)[i];
    float4 bv = ((const float4*)b)[c4];
    a.x = fmaxf(a.x + bv.x + xv.x, 0.f);
    a.y = fmaxf(a.y + bv.y + xv.y, 0.f);
    a.z = fmaxf(a.z + bv.z + xv.z, 0.f);
    a.w = fmaxf(a.w + bv.w + xv.w, 0.f);
    ((float4*)acc)[i] = a;
}

extern "C" void kernel_launch(void* const* d_in, const int* in_sizes, int n_in,
                              void* d_out, int out_size, void* d_ws, size_t ws_size,
                              hipStream_t stream)
{
    const float* x     = (const float*)d_in[0];
    const int*   ei    = (const int*)  d_in[1];   // [2,E] int32 (JAX x64-off)
    const float* h     = (const float*)d_in[3];
    const float* W     = (const float*)d_in[5];
    const float* b     = (const float*)d_in[6];
    const float* gamma = (const float*)d_in[7];
    const float* beta  = (const float*)d_in[8];

    const int N = in_sizes[4];        // batch has N elements
    const int E = in_sizes[1] / 2;

    // workspace layout
    float*          xw   = (float*)d_ws;                       // N*D f32
    float*          dinv = xw + (size_t)N * DD;                // N f32
    unsigned*       cnt  = (unsigned*)(dinv + N);              // N u32
    unsigned short* Wt   = (unsigned short*)(cnt + N);         // D*D bf16

    float* acc = (float*)d_out;                                // first N*D of output

    const int T = 256;
    zero_u32_kernel <<<(N + T - 1) / T, T, 0, stream>>>(cnt, N);
    degree_kernel   <<<(E + T - 1) / T, T, 0, stream>>>(ei + E, cnt, E);
    dinv_kernel     <<<(N + T - 1) / T, T, 0, stream>>>(cnt, dinv, N);
    packW_kernel    <<<(DD * DD + T - 1) / T, T, 0, stream>>>(W, Wt);

    ln_gemm_kernel  <<<(N + 15) / 16, T, 0, stream>>>(x, gamma, beta, Wt, xw, N);

    int nd4 = N * (DD / 4);
    selfloop_kernel <<<(nd4 + T - 1) / T, T, 0, stream>>>(xw, dinv, acc, N);

    long long sc_items = (long long)E * (DD / 4);
    scatter_kernel  <<<(unsigned)((sc_items + T - 1) / T), T, 0, stream>>>(ei, xw, dinv, acc, E);

    epilogue_kernel <<<(nd4 + T - 1) / T, T, 0, stream>>>(acc, x, b, N);

    // second output: pass-through of h
    hipMemcpyAsync((float*)d_out + (size_t)N * DD, h,
                   (size_t)N * DD * sizeof(float), hipMemcpyDeviceToDevice, stream);
}
